// GTLayer_3100966387738
// MI455X (gfx1250) — matrix operand order repaired and checked by reading, not compiled, not run on hardware
//
#include <hip/hip_runtime.h>

#define N_FEATS 9
#define VOCAB 119
#define HIDDEN 128
#define NUM_HEADS 8
#define HEAD_DIM 16
#define KNBR 16

typedef __attribute__((ext_vector_type(16))) __bf16 v16bf;
typedef __attribute__((ext_vector_type(8)))  float  v8f;

union Frag16 { v16bf v; unsigned u[8]; };

__device__ __forceinline__ unsigned short f2bf(float f) {
    unsigned u = __float_as_uint(f);
    u += 0x7FFFu + ((u >> 16) & 1u);            // round-to-nearest-even
    return (unsigned short)(u >> 16);
}
__device__ __forceinline__ float bflo(unsigned u) { return __uint_as_float(u << 16); }
__device__ __forceinline__ float bfhi(unsigned u) { return __uint_as_float(u & 0xffff0000u); }

// ---------------------------------------------------------------------------
// Kernel 0: convert Wq/Wk/Wv (f32, [k][col]) -> bf16 transposed [m][col][k]
// so that B-fragment (k,k+1) pairs are contiguous u32 loads.
// ---------------------------------------------------------------------------
__global__ __launch_bounds__(256) void prep_weights(const float* __restrict__ Wq,
                                                    const float* __restrict__ Wk,
                                                    const float* __restrict__ Wv,
                                                    unsigned short* __restrict__ wT) {
    int idx = blockIdx.x * 256 + threadIdx.x;   // 0 .. 3*128*128-1
    int m   = idx >> 14;
    int rem = idx & 16383;
    int col = rem >> 7;
    int k   = rem & 127;
    const float* W = (m == 0) ? Wq : ((m == 1) ? Wk : Wv);
    wT[idx] = f2bf(W[k * HIDDEN + col]);
}

// ---------------------------------------------------------------------------
// Kernel 1: atom-encoder + fused Q/K/V projections via v_wmma_f32_16x16x32_bf16
// Block = 128 threads (4 waves), 64 nodes per block (16-row tile per wave).
// Each wave owns LDS rows [wave*16, wave*16+16): used first for its A
// fragments, then reused as a store-transpose staging buffer so the 38 MB of
// q/k/v output goes out as coalesced global_store_b128 (no per-element guards).
// ---------------------------------------------------------------------------
__global__ __launch_bounds__(128) void encode_qkv(const int* __restrict__ X,
                                                  const float* __restrict__ emb,
                                                  const unsigned short* __restrict__ wT,
                                                  const float* __restrict__ bq,
                                                  const float* __restrict__ bk,
                                                  const float* __restrict__ bv,
                                                  unsigned short* __restrict__ qbf,
                                                  unsigned short* __restrict__ kbf,
                                                  unsigned short* __restrict__ vbf,
                                                  int N) {
    __shared__ unsigned short hsh[64 * HIDDEN];  // bf16 h tile / staging: 16 KB
    __shared__ int            xsh[64 * N_FEATS]; // staged feature ids

    const int tid       = threadIdx.x;           // 0..127
    const int base_node = blockIdx.x * 64;

    // stage feature indices
    for (int t = tid; t < 64 * N_FEATS; t += 128) {
        int g = base_node * N_FEATS + t;
        xsh[t] = (g < N * N_FEATS) ? X[g] : 0;
    }
    __syncthreads();

    // ---- stage 1: h[n][c] = sum_f emb[f][x[n][f]][c], store bf16 into LDS
    const int c = tid;                           // channel 0..127
    for (int i = 0; i < 64; ++i) {
        float s = 0.f;
        #pragma unroll
        for (int f = 0; f < N_FEATS; ++f) {
            int x = xsh[i * N_FEATS + f];
            s += emb[((size_t)f * VOCAB + x) * HIDDEN + c];
        }
        hsh[i * HIDDEN + c] = f2bf(s);
    }
    __syncthreads();                             // last barrier in the kernel

    // ---- stage 2: per-wave 16x128 @ 128x128 GEMMs with WMMA bf16
    const int wave = tid >> 5;
    const int lane = tid & 31;
    const int tile = blockIdx.x * 4 + wave;
    if (tile * 16 >= N) return;                  // whole wave idle; no barriers below

    const int half = lane >> 4;
    const int l15  = lane & 15;
    const int lrow = wave * 16 + (lane & 15);
    const int tilebase = tile * 16;
    unsigned short* stage = hsh + wave * 16 * HIDDEN;  // this wave's private slice

    // A fragments (16x32 bf16): VGPR j -> K = 8*half + 2j ; VGPR 4+j -> +16
    Frag16 a[4];
    #pragma unroll
    for (int ks = 0; ks < 4; ++ks) {
        int kb = ks * 32;
        #pragma unroll
        for (int j = 0; j < 4; ++j) {
            a[ks].u[j]     = *(const unsigned*)&hsh[lrow * HIDDEN + kb + 8 * half + 2 * j];
            a[ks].u[4 + j] = *(const unsigned*)&hsh[lrow * HIDDEN + kb + 16 + 8 * half + 2 * j];
        }
    }

    #pragma unroll
    for (int m = 0; m < 3; ++m) {
        const float* bias = (m == 0) ? bq : ((m == 1) ? bk : bv);
        unsigned short* dst = (m == 0) ? qbf : ((m == 1) ? kbf : vbf);
        #pragma unroll
        for (int ct = 0; ct < 8; ++ct) {
            const int colg = ct * 16 + l15;
            v8f acc = {};
            #pragma unroll
            for (int ks = 0; ks < 4; ++ks) {
                // B fragment (32x16 bf16): VGPR j -> K = 8*half + 2j ; VGPR 4+j -> +16
                Frag16 b;
                const unsigned short* wp = wT + ((size_t)(m * HIDDEN + colg)) * HIDDEN
                                              + ks * 32 + 8 * half;
                #pragma unroll
                for (int j = 0; j < 8; ++j)
                    b.u[j] = *(const unsigned*)&wp[2 * (j & 3) + 16 * (j >> 2)];
                acc = __builtin_amdgcn_wmma_f32_16x16x32_bf16(
                        false, a[ks].v, false, b.v, (short)0, acc, false, false);
            }
            // stage fragment into LDS (D: VGPR r -> row r + 8*half, col colg)
            const float bval = bias[colg];
            #pragma unroll
            for (int r = 0; r < 8; ++r) {
                float val = acc[r] + bval;
                if (m == 0) val *= 0.25f;        // q * HEAD_DIM^-0.5
                stage[(r + 8 * half) * HIDDEN + colg] = f2bf(val);
            }
        }
        // flush the 16x128 bf16 tile with coalesced b128 stores
        if (tilebase + 16 <= N) {
            const uint4* src  = (const uint4*)stage;               // 256 uint4
            uint4*       dstv = (uint4*)(dst + (size_t)tilebase * HIDDEN);
            #pragma unroll
            for (int i = 0; i < 8; ++i)
                dstv[i * 32 + lane] = src[i * 32 + lane];
        } else {                                 // partial tile (unused at N=50000)
            for (int i = lane; i < 16 * HIDDEN; i += 32) {
                int row = i >> 7;
                if (tilebase + row < N)
                    dst[(size_t)(tilebase + row) * HIDDEN + (i & 127)] = stage[i];
            }
        }
    }
}

// ---------------------------------------------------------------------------
// Kernel 2: ELL sparse attention, one wave32 per node.
// lane = nb*2 + hh : nb = neighbor slot (0..15), hh = head-half (heads hh*4..+3,
// i.e. dims hh*64..hh*64+63). Softmax over the 16 matching-parity lanes with
// shfl_xor(2,4,8,16); output via reduce-scatter butterfly so all 32 lanes end
// up owning 4 floats -> one coalesced float4 store per lane.
// ---------------------------------------------------------------------------
__global__ __launch_bounds__(256) void attn_kernel(const int* __restrict__ nbr_idx,
                                                   const int* __restrict__ nbr_mask,
                                                   const unsigned short* __restrict__ qbf,
                                                   const unsigned short* __restrict__ kbf,
                                                   const unsigned short* __restrict__ vbf,
                                                   float* __restrict__ out,
                                                   int N) {
    const int node = blockIdx.x * 8 + (threadIdx.x >> 5);
    if (node >= N) return;
    const int lane = threadIdx.x & 31;
    const int nb   = lane >> 1;
    const int hh   = lane & 1;

    const int idx = nbr_idx[node * KNBR + nb];
    const int msk = nbr_mask[node * KNBR + nb];

    const unsigned short* krow = kbf + (size_t)idx * HIDDEN + hh * 64;
    const unsigned short* vrow = vbf + (size_t)idx * HIDDEN + hh * 64;
    __builtin_prefetch(krow, 0, 3);              // global_prefetch_b8
    __builtin_prefetch(vrow, 0, 3);

    // load my 64 q values (bf16 -> f32)
    float qv[64];
    {
        const uint4* qp = (const uint4*)(qbf + (size_t)node * HIDDEN + hh * 64);
        #pragma unroll
        for (int t = 0; t < 8; ++t) {
            uint4 u = qp[t];
            qv[t*8+0] = bflo(u.x); qv[t*8+1] = bfhi(u.x);
            qv[t*8+2] = bflo(u.y); qv[t*8+3] = bfhi(u.y);
            qv[t*8+4] = bflo(u.z); qv[t*8+5] = bfhi(u.z);
            qv[t*8+6] = bflo(u.w); qv[t*8+7] = bfhi(u.w);
        }
    }

    // pull the k and v rows of neighbor `idx` into registers up front
    uint4 kk[8], vv[8];
    {
        const uint4* kp = (const uint4*)krow;
        const uint4* vp = (const uint4*)vrow;
        #pragma unroll
        for (int t = 0; t < 8; ++t) { kk[t] = kp[t]; vv[t] = vp[t]; }
    }

    // scores for my 4 heads (16 dims each)
    float s[4] = {0.f, 0.f, 0.f, 0.f};
    #pragma unroll
    for (int t = 0; t < 8; ++t) {                // head = t>>1
        uint4 u = kk[t];
        s[t >> 1] += qv[t*8+0]*bflo(u.x) + qv[t*8+1]*bfhi(u.x)
                   + qv[t*8+2]*bflo(u.y) + qv[t*8+3]*bfhi(u.y)
                   + qv[t*8+4]*bflo(u.z) + qv[t*8+5]*bfhi(u.z)
                   + qv[t*8+6]*bflo(u.w) + qv[t*8+7]*bfhi(u.w);
    }
    #pragma unroll
    for (int t = 0; t < 4; ++t) s[t] = msk ? s[t] : -1e9f;

    // softmax over the 16 neighbor lanes (same parity): xor 2,4,8,16
    float aw[4];
    #pragma unroll
    for (int t = 0; t < 4; ++t) {
        float mx = s[t];
        #pragma unroll
        for (int off = 2; off < 32; off <<= 1) mx = fmaxf(mx, __shfl_xor(mx, off, 32));
        float e = __expf(s[t] - mx);
        float sum = e;
        #pragma unroll
        for (int off = 2; off < 32; off <<= 1) sum += __shfl_xor(sum, off, 32);
        aw[t] = e / sum;
    }

    // weighted V accumulate (per-lane)
    float o[64];
    #pragma unroll
    for (int t = 0; t < 8; ++t) {
        uint4 u = vv[t];
        float w = aw[t >> 1];
        o[t*8+0] = w * bflo(u.x); o[t*8+1] = w * bfhi(u.x);
        o[t*8+2] = w * bflo(u.y); o[t*8+3] = w * bfhi(u.y);
        o[t*8+4] = w * bflo(u.z); o[t*8+5] = w * bfhi(u.z);
        o[t*8+6] = w * bflo(u.w); o[t*8+7] = w * bfhi(u.w);
    }

    // reduce-scatter butterfly over lane bits 1..4: each step halves the
    // owned segment; lane keeps lower half if its step-bit is 0.
    #pragma unroll
    for (int st = 1; st <= 4; ++st) {
        const int seg = 64 >> st;
        const bool hi = (lane >> st) & 1;
        #pragma unroll
        for (int j = 0; j < seg; ++j) {
            float send = hi ? o[j]       : o[j + seg];  // half I discard
            float keep = hi ? o[j + seg] : o[j];
            o[j] = keep + __shfl_xor(send, 1 << st, 32);
        }
    }
    // lane owns 4 floats at segment base determined by its bits
    const int base = ((lane >> 1) & 1) * 32 + ((lane >> 2) & 1) * 16
                   + ((lane >> 3) & 1) * 8  + ((lane >> 4) & 1) * 4;
    float4* dst = (float4*)(out + (size_t)node * HIDDEN + hh * 64 + base);
    *dst = make_float4(o[0], o[1], o[2], o[3]);
}

// ---------------------------------------------------------------------------
extern "C" void kernel_launch(void* const* d_in, const int* in_sizes, int n_in,
                              void* d_out, int out_size, void* d_ws, size_t ws_size,
                              hipStream_t stream) {
    const int*   X        = (const int*)d_in[0];
    const int*   nbr_idx  = (const int*)d_in[1];
    const int*   nbr_mask = (const int*)d_in[2];
    const float* emb      = (const float*)d_in[3];
    const float* Wq       = (const float*)d_in[4];
    const float* bq       = (const float*)d_in[5];
    const float* Wk       = (const float*)d_in[6];
    const float* bk       = (const float*)d_in[7];
    const float* Wv       = (const float*)d_in[8];
    const float* bv       = (const float*)d_in[9];

    const int N = in_sizes[0] / N_FEATS;         // 50000

    // workspace layout (bf16): wT[3*128*128] | q[N*128] | k[N*128] | v[N*128]
    unsigned short* ws16 = (unsigned short*)d_ws;
    unsigned short* wT   = ws16;
    unsigned short* qbf  = wT  + 3 * HIDDEN * HIDDEN;
    unsigned short* kbf  = qbf + (size_t)N * HIDDEN;
    unsigned short* vbf  = kbf + (size_t)N * HIDDEN;

    prep_weights<<<(3 * HIDDEN * HIDDEN) / 256, 256, 0, stream>>>(Wq, Wk, Wv, wT);

    encode_qkv<<<(N + 63) / 64, 128, 0, stream>>>(X, emb, wT, bq, bk, bv,
                                                  qbf, kbf, vbf, N);

    attn_kernel<<<(N + 7) / 8, 256, 0, stream>>>(nbr_idx, nbr_mask, qbf, kbf, vbf,
                                                 (float*)d_out, N);
}
